// Encoder_16750372454478
// MI455X (gfx1250) — matrix operand order repaired and checked by reading, not compiled, not run on hardware
//
#include <hip/hip_runtime.h>

// ---------------------------------------------------------------- constants
#define B_  8
#define S_  1024
#define D_  1024
#define H_  16
#define DK_ 64
#define L_  4
#define EPS_ 1e-5f

typedef __bf16 bf16;
typedef __attribute__((ext_vector_type(16))) __bf16 v16bf;
typedef __attribute__((ext_vector_type(8)))  __bf16 v8bf;
typedef __attribute__((ext_vector_type(8)))  float  v8f;
typedef unsigned short ushort_t;

#define DEV static __device__ __forceinline__

DEV ushort_t f2bf(float f) {
    union { float f; unsigned u; } c; c.f = f;
    unsigned u = c.u + 0x7FFFu + ((c.u >> 16) & 1u);   // round-to-nearest-even
    return (ushort_t)(u >> 16);
}

// A-operand fragment (16x32, MxK): lane holds row (lane&15);
// element j -> K = (j<8 ? j : j+8) + (hi ? 8 : 0)
DEV v16bf frag_a(const ushort_t* base, int stride, int lane) {
    const int hi = (lane >> 4) & 1;
    const bf16* p = (const bf16*)base + (size_t)(lane & 15) * stride + hi * 8;
    v8bf lo = *(const v8bf*)p;
    v8bf hg = *(const v8bf*)(p + 16);
    return __builtin_shufflevector(lo, hg, 0,1,2,3,4,5,6,7,8,9,10,11,12,13,14,15);
}

// B-operand fragment (32x16, KxN) gathered from an [N][K] staged tile:
// lane holds column n = lane&15; element j -> K = (j<8 ? j : j+8) + (hi ? 8 : 0)
DEV v16bf frag_b(const ushort_t* base, int stride, int lane) {
    const int hi = (lane >> 4) & 1;
    const bf16* p = (const bf16*)base + (size_t)(lane & 15) * stride + hi * 8;
    v8bf lo = *(const v8bf*)p;
    v8bf hg = *(const v8bf*)(p + 16);
    return __builtin_shufflevector(lo, hg, 0,1,2,3,4,5,6,7,8,9,10,11,12,13,14,15);
}

DEV v8f wmma_bf16(v16bf a, v16bf b, v8f c) {
    return __builtin_amdgcn_wmma_f32_16x16x32_bf16(false, a, false, b, (short)0, c,
                                                   false, false);
}

DEV unsigned lds_off(const void* p) {   // low 32 bits of a flat LDS address
    return (unsigned)(unsigned long long)p;
}

// ---------------------------------------------------------------- GEMM
// C[M,N] = act( A[M,K] * W[N,K]^T [+ bias] [+ res] )   (A,W bf16; accum f32)
// Double-buffered LDS tiles filled by GLOBAL_LOAD_ASYNC_TO_LDS_B128 DMA.
#define GTM 128
#define GTN 128
#define GTK 32
#define GSTR 40   // padded LDS stride (halves); 80B = multiple of 16B

template<bool BIAS, bool RES, bool OUTF, bool OUTB, bool RELU>
__global__ __launch_bounds__(256) void gemm_bf16_kernel(
    const ushort_t* __restrict__ A, const ushort_t* __restrict__ W,
    const float* __restrict__ bias, const float* __restrict__ res,
    float* __restrict__ outF, ushort_t* __restrict__ outB,
    int M, int N, int K)
{
    __shared__ ushort_t sA[2][GTM * GSTR];   // 2 x 10KB
    __shared__ ushort_t sB[2][GTN * GSTR];   // 2 x 10KB   (total 40KB)

    const int tid  = threadIdx.x;
    const int lane = tid & 31;
    const int wid  = tid >> 5;       // 0..7
    const int wm   = wid & 1;        // 2 waves over M
    const int wn   = wid >> 1;       // 4 waves over N
    const int bm   = blockIdx.y * GTM;
    const int bn   = blockIdx.x * GTN;

    v8f acc[4][2] = {};

    // staging map: 256 threads x 32B cover a 128x32 bf16 tile per matrix
    const int r = tid >> 1;            // 0..127
    const int c = (tid & 1) << 4;      // 0 or 16 (halves)
    const ushort_t* gA = A + (size_t)(bm + r) * K + c;
    const ushort_t* gW = W + (size_t)(bn + r) * K + c;
    const unsigned lA0 = lds_off(&sA[0][r * GSTR + c]);
    const unsigned lA1 = lds_off(&sA[1][r * GSTR + c]);
    const unsigned lB0 = lds_off(&sB[0][r * GSTR + c]);
    const unsigned lB1 = lds_off(&sB[1][r * GSTR + c]);

    // Async DMA of one 128x32 tile of A and W into LDS buffer `buf`.
    // 4 instructions per wave -> ASYNCcnt += 4.
    auto issue = [&](int k0, int buf) {
        unsigned long long ga = (unsigned long long)(gA + k0);
        unsigned long long gw = (unsigned long long)(gW + k0);
        unsigned la = buf ? lA1 : lA0;
        unsigned lb = buf ? lB1 : lB0;
        asm volatile(
            "global_load_async_to_lds_b128 %0, %2, off\n\t"
            "global_load_async_to_lds_b128 %0, %2, off offset:16\n\t"
            "global_load_async_to_lds_b128 %1, %3, off\n\t"
            "global_load_async_to_lds_b128 %1, %3, off offset:16"
            :: "v"(la), "v"(lb), "v"(ga), "v"(gw) : "memory");
    };

    issue(0, 0);
    int cur = 0;
    for (int k0 = 0; k0 < K; k0 += GTK) {
        if (k0 + GTK < K) {
            issue(k0 + GTK, cur ^ 1);                       // prefetch next tile
            asm volatile("s_wait_asynccnt 0x4" ::: "memory"); // tile k0 resident
        } else {
            asm volatile("s_wait_asynccnt 0x0" ::: "memory");
        }
        __syncthreads();   // tile k0 visible to all waves

        const ushort_t* bA = &sA[cur][0];
        const ushort_t* bW = &sB[cur][0];
        v16bf af[4], bfm[2];
        #pragma unroll
        for (int i = 0; i < 4; i++)
            af[i] = frag_a(bA + (wm * 64 + i * 16) * GSTR, GSTR, lane);
        #pragma unroll
        for (int j = 0; j < 2; j++)
            bfm[j] = frag_b(bW + (wn * 32 + j * 16) * GSTR, GSTR, lane);
        #pragma unroll
        for (int i = 0; i < 4; i++)
            #pragma unroll
            for (int j = 0; j < 2; j++)
                acc[i][j] = wmma_bf16(af[i], bfm[j], acc[i][j]);

        __syncthreads();   // all waves done reading buf `cur` before its reuse
        cur ^= 1;
    }

    const int hi = (lane >> 4) & 1;
    const int cl = lane & 15;
    #pragma unroll
    for (int i = 0; i < 4; i++) {
        #pragma unroll
        for (int j = 0; j < 2; j++) {
            const int row0 = bm + wm * 64 + i * 16 + hi * 8;
            const int coln = bn + wn * 32 + j * 16 + cl;
            float bv = 0.f;
            if constexpr (BIAS) bv = bias[coln];
            #pragma unroll
            for (int rr = 0; rr < 8; rr++) {
                const int row = row0 + rr;
                float v = acc[i][j][rr] + bv;
                const size_t idx = (size_t)row * N + coln;
                if constexpr (RES)  v += res[idx];
                if constexpr (RELU) v = v > 0.f ? v : 0.f;
                if constexpr (OUTF) outF[idx] = v;
                if constexpr (OUTB) outB[idx] = f2bf(v);
            }
        }
    }
}

// ---------------------------------------------------------------- flash attention
// Q,K,V,O layout: [B, S, H*DK]  (head h at column offset h*DK)
#define FKSTR 72
#define FVSTR 40
#define FPSTR 40

__global__ __launch_bounds__(256) void flash_attn_kernel(
    const ushort_t* __restrict__ Q, const ushort_t* __restrict__ Kp,
    const ushort_t* __restrict__ Vp, ushort_t* __restrict__ O)
{
    __shared__ ushort_t sK[32 * FKSTR];        // [key][dk]
    __shared__ ushort_t sV[64 * FVSTR];        // transposed: [dk][key]
    __shared__ ushort_t sP[8 * 16 * FPSTR];    // per-wave P tiles [16][32]

    const int tid  = threadIdx.x;
    const int lane = tid & 31;
    const int wid  = tid >> 5;
    const int hi   = (lane >> 4) & 1;
    const int cl   = lane & 15;
    const int bh   = blockIdx.y;               // b*H + h
    const int b    = bh >> 4, h = bh & 15;
    const int q0   = blockIdx.x * 128 + wid * 16;

    const size_t headoff = (size_t)h * DK_;
    const ushort_t* qbase = Q + ((size_t)b * S_ + q0) * D_ + headoff;

    v16bf qf0 = frag_a(qbase,      D_, lane);   // dk 0..31
    v16bf qf1 = frag_a(qbase + 32, D_, lane);   // dk 32..63

    v8f oacc[4] = {};
    float mrow[8], lrow[8];
    #pragma unroll
    for (int rr = 0; rr < 8; rr++) { mrow[rr] = -3.0e38f; lrow[rr] = 0.f; }

    ushort_t* sPw = &sP[wid * 16 * FPSTR];

    for (int kk = 0; kk < S_; kk += 32) {
        __syncthreads();
        {   // stage K (32x64) and transposed V
            const int rr = tid >> 3, cc = (tid & 7) << 3;
            const ushort_t* gk = Kp + ((size_t)b * S_ + kk + rr) * D_ + headoff + cc;
            *(v8bf*)&sK[rr * FKSTR + cc] = *(const v8bf*)gk;
            const ushort_t* gv = Vp + ((size_t)b * S_ + kk + rr) * D_ + headoff + cc;
            v8bf vv = *(const v8bf*)gv;
            #pragma unroll
            for (int e = 0; e < 8; e++)
                sV[(cc + e) * FVSTR + rr] = __builtin_bit_cast(ushort_t, (bf16)vv[e]);
        }
        __syncthreads();

        // scores: S = Q * K^T   (M=16 queries, N=32 keys, K=64 contraction)
        v8f s0 = {}, s1 = {};
        {
            v16bf kb00 = frag_b(&sK[0],               FKSTR, lane);
            v16bf kb01 = frag_b(&sK[32],              FKSTR, lane);
            v16bf kb10 = frag_b(&sK[16 * FKSTR],      FKSTR, lane);
            v16bf kb11 = frag_b(&sK[16 * FKSTR + 32], FKSTR, lane);
            s0 = wmma_bf16(qf0, kb00, s0); s0 = wmma_bf16(qf1, kb01, s0);
            s1 = wmma_bf16(qf0, kb10, s1); s1 = wmma_bf16(qf1, kb11, s1);
        }

        // online softmax (rows live in 16-lane halves of the C layout)
        #pragma unroll
        for (int rr = 0; rr < 8; rr++) {
            float a0 = s0[rr] * 0.125f, a1 = s1[rr] * 0.125f;  // 1/sqrt(64)
            float mb = fmaxf(a0, a1);
            #pragma unroll
            for (int off = 1; off < 16; off <<= 1)
                mb = fmaxf(mb, __shfl_xor(mb, off, 32));
            const float mn   = fmaxf(mrow[rr], mb);
            const float corr = __expf(mrow[rr] - mn);
            const float p0 = __expf(a0 - mn), p1 = __expf(a1 - mn);
            float ps = p0 + p1;
            #pragma unroll
            for (int off = 1; off < 16; off <<= 1)
                ps += __shfl_xor(ps, off, 32);
            lrow[rr] = lrow[rr] * corr + ps;
            mrow[rr] = mn;
            #pragma unroll
            for (int t = 0; t < 4; t++) oacc[t][rr] *= corr;
            const int prow = rr + hi * 8;
            sPw[prow * FPSTR + cl]      = f2bf(p0);
            sPw[prow * FPSTR + 16 + cl] = f2bf(p1);
        }

        // O += P * V   (M=16, N=64 dk, K=32 keys)
        v16bf pf = frag_a(sPw, FPSTR, lane);
        #pragma unroll
        for (int t = 0; t < 4; t++) {
            v16bf vb = frag_b(&sV[t * 16 * FVSTR], FVSTR, lane);
            oacc[t] = wmma_bf16(pf, vb, oacc[t]);
        }
    }

    #pragma unroll
    for (int t = 0; t < 4; t++) {
        #pragma unroll
        for (int rr = 0; rr < 8; rr++) {
            const int srow = q0 + rr + hi * 8;
            const float v = oacc[t][rr] / lrow[rr];
            const size_t idx = ((size_t)b * S_ + srow) * D_ + headoff + t * 16 + cl;
            O[idx] = f2bf(v);
        }
    }
}

// ---------------------------------------------------------------- LayerNorm over (S,D)
__global__ __launch_bounds__(32) void ln_init_kernel(float* stats) {
    if (threadIdx.x < 2 * B_) stats[threadIdx.x] = 0.f;
}

__global__ __launch_bounds__(256) void ln_stats_kernel(
    const float* __restrict__ x, float* __restrict__ stats)
{
    __shared__ float s1[256], s2[256];
    const int b = blockIdx.x >> 5, blk = blockIdx.x & 31;
    const size_t base = (size_t)b * ((size_t)S_ * D_);
    float a = 0.f, q = 0.f;
    for (size_t i = (size_t)blk * 256 + threadIdx.x; i < (size_t)S_ * D_; i += 32 * 256) {
        const float v = x[base + i];
        a += v; q += v * v;
    }
    s1[threadIdx.x] = a; s2[threadIdx.x] = q;
    __syncthreads();
    for (int o = 128; o > 0; o >>= 1) {
        if ((int)threadIdx.x < o) {
            s1[threadIdx.x] += s1[threadIdx.x + o];
            s2[threadIdx.x] += s2[threadIdx.x + o];
        }
        __syncthreads();
    }
    if (threadIdx.x == 0) {
        atomicAdd(&stats[2 * b],     s1[0]);
        atomicAdd(&stats[2 * b + 1], s2[0]);
    }
}

__global__ __launch_bounds__(256) void ln_apply_kernel(
    const float* __restrict__ x, const float* __restrict__ stats,
    ushort_t* __restrict__ outB)
{
    const size_t idx = (size_t)blockIdx.x * 256 + threadIdx.x;
    const int b = (int)(idx / ((size_t)S_ * D_));
    const float inv = 1.f / (float)((size_t)S_ * D_);
    const float m  = stats[2 * b] * inv;
    const float va = stats[2 * b + 1] * inv - m * m;
    const float rs = rsqrtf(va + EPS_);
    outB[idx] = f2bf((x[idx] - m) * rs);
}

// ---------------------------------------------------------------- misc elementwise
__global__ __launch_bounds__(256) void posenc_kernel(
    const float* __restrict__ X, float* __restrict__ xf, ushort_t* __restrict__ xb)
{
    const size_t idx = (size_t)blockIdx.x * 256 + threadIdx.x;
    const int d = (int)(idx % D_);
    const int s = (int)((idx / D_) % S_);
    const float expo = (float)(2 * (d >> 1)) / (float)D_;
    const float ang  = (float)s * __powf(1000.0f, -expo);
    const float pe   = (d & 1) ? __cosf(ang) : __sinf(ang);
    const float v = X[idx] + pe;
    xf[idx] = v;
    xb[idx] = f2bf(v);
}

__global__ __launch_bounds__(256) void cvt_kernel(
    const float* __restrict__ in, ushort_t* __restrict__ out, size_t n)
{
    const size_t idx = (size_t)blockIdx.x * 256 + threadIdx.x;
    if (idx < n) out[idx] = f2bf(in[idx]);
}

// ---------------------------------------------------------------- launch
extern "C" void kernel_launch(void* const* d_in, const int* in_sizes, int n_in,
                              void* d_out, int out_size, void* d_ws, size_t ws_size,
                              hipStream_t stream)
{
    (void)in_sizes; (void)n_in; (void)out_size; (void)ws_size;
    const float* X     = (const float*)d_in[0];
    const float* Wq    = (const float*)d_in[1];
    const float* Wk    = (const float*)d_in[2];
    const float* Wv    = (const float*)d_in[3];
    const float* W0    = (const float*)d_in[4];
    const float* Wr_w  = (const float*)d_in[5];
    const float* Wr_b  = (const float*)d_in[6];
    const float* Wf0_w = (const float*)d_in[7];
    const float* Wf0_b = (const float*)d_in[8];
    const float* Wf1_w = (const float*)d_in[9];
    const float* Wf1_b = (const float*)d_in[10];

    const size_t DD  = (size_t)D_ * D_;          // 1M
    const size_t WN  = (size_t)L_ * DD;          // 4M per weight tensor
    const size_t ACT = (size_t)B_ * S_ * D_;     // 8M

    char* ws = (char*)d_ws;
    size_t off = 0;
    auto take = [&](size_t bytes) -> char* {
        char* p = ws + off; off += (bytes + 255) & ~(size_t)255; return p;
    };
    ushort_t* wq_b  = (ushort_t*)take(WN * 2);
    ushort_t* wk_b  = (ushort_t*)take(WN * 2);
    ushort_t* wv_b  = (ushort_t*)take(WN * 2);
    ushort_t* w0_b  = (ushort_t*)take(WN * 2);
    ushort_t* wrw_b = (ushort_t*)take(WN * 2);
    ushort_t* wf0_b = (ushort_t*)take(WN * 2);
    ushort_t* wf1_b = (ushort_t*)take(WN * 2);
    float*    xf    = (float*)   take(ACT * 4);
    ushort_t* xb    = (ushort_t*)take(ACT * 2);
    ushort_t* qb    = (ushort_t*)take(ACT * 2);
    ushort_t* kb    = (ushort_t*)take(ACT * 2);
    ushort_t* vb    = (ushort_t*)take(ACT * 2);
    float*    t0f   = (float*)   take(ACT * 4);
    ushort_t* ln1b  = (ushort_t*)take(ACT * 2);
    float*    r0f   = (float*)   take(ACT * 4);
    ushort_t* r0b   = (ushort_t*)take(ACT * 2);
    float*    stats = (float*)   take(256);
    // aliases (buffers dead by the time the alias is written)
    ushort_t* ob    = (ushort_t*)r0f;   // attn output; overwritten later by r0f
    float*    t5f   = t0f;              // ff+r0; t0f dead after ln1
    ushort_t* ln2b  = ln1b;             // ln1b dead after r0 GEMM
    ushort_t* ff1b  = xb;               // xb dead after QKV GEMMs

    const dim3 blk(256);
    const unsigned actBlocks = (unsigned)(ACT / 256);
    const unsigned wBlocks   = (unsigned)((WN + 255) / 256);
    const dim3 ggrid(D_ / GTN, (B_ * S_) / GTM);   // (8, 64)
    const dim3 fgrid(S_ / 128, B_ * H_);           // (8, 128)
    const int M = B_ * S_, N = D_, K = D_;

    const float* wsrc[7] = {Wq, Wk, Wv, W0, Wr_w, Wf0_w, Wf1_w};
    ushort_t*    wdst[7] = {wq_b, wk_b, wv_b, w0_b, wrw_b, wf0_b, wf1_b};
    for (int i = 0; i < 7; i++)
        cvt_kernel<<<wBlocks, blk, 0, stream>>>(wsrc[i], wdst[i], WN);

    posenc_kernel<<<actBlocks, blk, 0, stream>>>(X, xf, xb);

    for (int l = 0; l < L_; l++) {
        const ushort_t* wq_l  = wq_b  + (size_t)l * DD;
        const ushort_t* wk_l  = wk_b  + (size_t)l * DD;
        const ushort_t* wv_l  = wv_b  + (size_t)l * DD;
        const ushort_t* w0_l  = w0_b  + (size_t)l * DD;
        const ushort_t* wrw_l = wrw_b + (size_t)l * DD;
        const ushort_t* wf0_l = wf0_b + (size_t)l * DD;
        const ushort_t* wf1_l = wf1_b + (size_t)l * DD;
        const float* wrb_l  = Wr_b  + (size_t)l * D_;
        const float* wf0b_l = Wf0_b + (size_t)l * D_;
        const float* wf1b_l = Wf1_b + (size_t)l * D_;

        // QKV projections (bf16 out only)
        gemm_bf16_kernel<false,false,false,true,false><<<ggrid, blk, 0, stream>>>(
            xb, wq_l, nullptr, nullptr, nullptr, qb, M, N, K);
        gemm_bf16_kernel<false,false,false,true,false><<<ggrid, blk, 0, stream>>>(
            xb, wk_l, nullptr, nullptr, nullptr, kb, M, N, K);
        gemm_bf16_kernel<false,false,false,true,false><<<ggrid, blk, 0, stream>>>(
            xb, wv_l, nullptr, nullptr, nullptr, vb, M, N, K);
        // attention
        flash_attn_kernel<<<fgrid, blk, 0, stream>>>(qb, kb, vb, ob);
        // mha = O @ W0^T, fused residual t0 = x + mha (f32)
        gemm_bf16_kernel<false,true,true,false,false><<<ggrid, blk, 0, stream>>>(
            ob, w0_l, nullptr, xf, t0f, nullptr, M, N, K);
        // ln1 = LN(t0)
        ln_init_kernel<<<1, 32, 0, stream>>>(stats);
        ln_stats_kernel<<<B_ * 32, blk, 0, stream>>>(t0f, stats);
        ln_apply_kernel<<<actBlocks, blk, 0, stream>>>(t0f, stats, ln1b);
        // r0 = ln1 @ Wr^T + br   (f32 + bf16 outputs)
        gemm_bf16_kernel<true,false,true,true,false><<<ggrid, blk, 0, stream>>>(
            ln1b, wrw_l, wrb_l, nullptr, r0f, r0b, M, N, K);
        // ff1 = relu(r0 @ Wf0^T + b0)
        gemm_bf16_kernel<true,false,false,true,true><<<ggrid, blk, 0, stream>>>(
            r0b, wf0_l, wf0b_l, nullptr, nullptr, ff1b, M, N, K);
        // t5 = ff1 @ Wf1^T + b1 + r0
        gemm_bf16_kernel<true,true,true,false,false><<<ggrid, blk, 0, stream>>>(
            ff1b, wf1_l, wf1b_l, r0f, t5f, nullptr, M, N, K);
        // ln2 = LN(t5)
        ln_init_kernel<<<1, 32, 0, stream>>>(stats);
        ln_stats_kernel<<<B_ * 32, blk, 0, stream>>>(t5f, stats);
        ln_apply_kernel<<<actBlocks, blk, 0, stream>>>(t5f, stats, ln2b);
        // x = ln2 @ Wr^T + br   (f32 for next-layer residual, bf16 for next GEMMs)
        gemm_bf16_kernel<true,false,true,true,false><<<ggrid, blk, 0, stream>>>(
            ln2b, wrw_l, wrb_l, nullptr, xf, xb, M, N, K);
    }

    hipMemcpyAsync(d_out, xf, ACT * sizeof(float), hipMemcpyDeviceToDevice, stream);
}
